// PDEFunc_29059748725443
// MI455X (gfx1250) — matrix operand order repaired and checked by reading, not compiled, not run on hardware
//
#include <hip/hip_runtime.h>
#include <hip/hip_bf16.h>

// ---------------------------------------------------------------------------
// Tsit5 PDE solve restructured for MI455X (gfx1250, wave32, WMMA):
//   f1 = init_mlp(x) + DT * sum_{n,s} b_s * g(t_{n,s})
// Linearity of grad_mlp layer0/layer3 lets us hoist them out of the 600
// stage evaluations:
//   u0 = x @ gW0^T  (once);  per eval h0 = relu(sv*u0 + gb0)
//   A  = sum_ev b_ev * relu(relu(h0@W1^T+gb1)@W2^T+gb2)   (phase B, fused)
//   out[b,i] = f0[b,i] + sum_j x[b,j]*(DT*Q[b,i*64+j] + gb3[i*64+j]),
//     Q = A @ gW3^T   (gW3 flat index (i*64+j)*256+k == i*16384 + j*256 + k)
// All GEMMs: f16 inputs, f32 accumulate via v_wmma_f32_16x16x32_f16.
// ---------------------------------------------------------------------------

typedef __attribute__((ext_vector_type(16))) _Float16 v16h;
typedef __attribute__((ext_vector_type(8)))  _Float16 v8h;
typedef __attribute__((ext_vector_type(4)))  _Float16 v4h;
typedef __attribute__((ext_vector_type(8)))  float    v8f;

__device__ __constant__ float kTsitC[6] = {0.0f, 0.161f, 0.327f, 0.9f,
                                           0.9800255409045097f, 1.0f};
__device__ __constant__ float kTsitB[6] = {0.09646076681806523f, 0.01f,
                                           0.4798896504144996f, 1.379008574103742f,
                                           -3.290069515436081f, 2.324710524099774f};

__device__ __forceinline__ v8f vzero8() {
  v8f z;
#pragma unroll
  for (int i = 0; i < 8; ++i) z[i] = 0.0f;
  return z;
}

// A-fragment (16x32 f16): lane L holds row L%16; halves at K = 8*(L/16)+[0..8)
// and K = 16+8*(L/16)+[0..8)  -> two 16B chunks, 32B apart.
__device__ __forceinline__ v16h ld_fragA(const _Float16* p) {
  v8h lo = *(const v8h*)(p);
  v8h hi = *(const v8h*)(p + 16);
  v16h r;
#pragma unroll
  for (int i = 0; i < 8; ++i) { r[i] = lo[i]; r[i + 8] = hi[i]; }
  return r;
}

// B-fragment (32x16 f16): lane L holds col N=L%16; halves at K = 8*(L/16)+[0..8)
// and K = 16+8*(L/16)+[0..8) of weight row N (B[k][n] = W[n][k]).
__device__ __forceinline__ v16h ld_fragB(const _Float16* p) {
  v8h lo = *(const v8h*)(p);
  v8h hi = *(const v8h*)(p + 16);
  v16h r;
#pragma unroll
  for (int i = 0; i < 8; ++i) { r[i] = lo[i]; r[i + 8] = hi[i]; }
  return r;
}

__device__ __forceinline__ v8f wmma16(v16h a, v16h b, v8f c) {
  return __builtin_amdgcn_wmma_f32_16x16x32_f16(false, a, false, b,
                                                (short)0, c, false, false);
}

// ---------------------------------------------------------------------------
// f32 -> f16 convert
__global__ void cvt_kernel(const float* __restrict__ s, _Float16* __restrict__ d,
                           int n) {
  int i = blockIdx.x * blockDim.x + threadIdx.x;
  if (i < n) d[i] = (_Float16)s[i];
}

// ---------------------------------------------------------------------------
// Generic 64x64-tile WMMA GEMM: out = act( A[M,K] @ W[N,K]^T + bias )
// Block 128 = 4 waves (2x2), wave tile 32x32 (2x2 WMMA tiles). K,N mult of 64.
// EPI: 0 = relu+bias -> f16, 1 = bias -> f32, 2 = raw -> f32
template <int EPI>
__global__ __launch_bounds__(128) void gemm_kernel(
    const _Float16* __restrict__ A, const _Float16* __restrict__ W,
    const float* __restrict__ bias, float* __restrict__ outf,
    _Float16* __restrict__ outh, int N, int K) {
  __shared__ _Float16 sA[64 * 72];
  __shared__ _Float16 sW[64 * 72];
  const int tid = threadIdx.x;
  const int bm = blockIdx.x * 64, bn = blockIdx.y * 64;
  const int lane = tid & 31, wave = tid >> 5;
  const int wm = wave >> 1, wn = wave & 1;
  const int lmod = lane & 15, lhalf = lane >> 4;

  v8f d[2][2];
#pragma unroll
  for (int mt = 0; mt < 2; ++mt)
#pragma unroll
    for (int nt = 0; nt < 2; ++nt) d[mt][nt] = vzero8();

  for (int kc = 0; kc < K; kc += 64) {
    for (int i = tid; i < 512; i += 128) {
      const int r = i >> 3, c = i & 7;
      *(uint4*)(sA + r * 72 + c * 8) =
          *(const uint4*)(A + (size_t)(bm + r) * K + kc + c * 8);
      *(uint4*)(sW + r * 72 + c * 8) =
          *(const uint4*)(W + (size_t)(bn + r) * K + kc + c * 8);
    }
    __syncthreads();
#pragma unroll
    for (int ks = 0; ks < 2; ++ks) {
      v16h af[2], bf[2];
#pragma unroll
      for (int mt = 0; mt < 2; ++mt)
        af[mt] = ld_fragA(sA + (wm * 32 + mt * 16 + lmod) * 72 + ks * 32 + lhalf * 8);
#pragma unroll
      for (int nt = 0; nt < 2; ++nt)
        bf[nt] = ld_fragB(sW + (wn * 32 + nt * 16 + lmod) * 72 + ks * 32 + lhalf * 8);
#pragma unroll
      for (int mt = 0; mt < 2; ++mt)
#pragma unroll
        for (int nt = 0; nt < 2; ++nt) d[mt][nt] = wmma16(af[mt], bf[nt], d[mt][nt]);
    }
    __syncthreads();
  }
#pragma unroll
  for (int mt = 0; mt < 2; ++mt)
#pragma unroll
    for (int nt = 0; nt < 2; ++nt) {
      const int col = bn + wn * 32 + nt * 16 + lmod;
      const float b = (EPI == 2) ? 0.0f : bias[col];
#pragma unroll
      for (int v = 0; v < 8; ++v) {
        const int row = bm + wm * 32 + mt * 16 + v + 8 * lhalf;
        const float val = d[mt][nt][v] + b;
        if (EPI == 0)
          outh[(size_t)row * N + col] = (_Float16)fmaxf(val, 0.0f);
        else
          outf[(size_t)row * N + col] = val;
      }
    }
}

// ---------------------------------------------------------------------------
// Phase B: for one 64-row tile and one eval-chunk, accumulate
//   Aacc = sum_ev b_ev * relu( relu( relu(sv*u0+gb0) @W1^T +gb1 ) @W2^T +gb2 )
// W1,W2 resident in LDS (f16, stride 264 halves = conflict-free fragments);
// single activation buffer reused h0->h1 (GEMM1 result parked in registers).
// Block 256 = 8 waves (2x4), wave tile 32x64 (2x4 WMMA tiles).
__global__ __launch_bounds__(256) void phaseB_kernel(
    const float* __restrict__ u0, const float* __restrict__ gb0,
    const float* __restrict__ gb1, const float* __restrict__ gb2,
    const _Float16* __restrict__ W1h, const _Float16* __restrict__ W2h,
    float* __restrict__ Apart, int CS) {
  extern __shared__ char smemraw[];
  _Float16* sW1 = (_Float16*)smemraw;          // 256*264 halves
  _Float16* sW2 = sW1 + 256 * 264;             // 256*264 halves
  _Float16* sact = sW2 + 256 * 264;            // 64*264 halves
  float* sb0 = (float*)(sact + 64 * 264);      // 256 f32 each
  float* sb1 = sb0 + 256;
  float* sb2 = sb1 + 256;

  const int tid = threadIdx.x;
  const int row0 = blockIdx.x * 64;
  const int chunk = blockIdx.y;

  for (int i = tid; i < 8192; i += 256) {  // 256*256 halves / 8 per uint4
    const int r = i >> 5, c = i & 31;
    *(uint4*)(sW1 + r * 264 + c * 8) = ((const uint4*)W1h)[i];
    *(uint4*)(sW2 + r * 264 + c * 8) = ((const uint4*)W2h)[i];
  }
  sb0[tid] = gb0[tid];
  sb1[tid] = gb1[tid];
  sb2[tid] = gb2[tid];
  __syncthreads();

  const int lane = tid & 31, wave = tid >> 5;
  const int wm = wave >> 2, wn = wave & 3;
  const int lmod = lane & 15, lhalf = lane >> 4;

  v8f aacc[2][4];
#pragma unroll
  for (int mt = 0; mt < 2; ++mt)
#pragma unroll
    for (int nt = 0; nt < 4; ++nt) aacc[mt][nt] = vzero8();

  const int nev = CS * 6;
  for (int ev = 0; ev < nev; ++ev) {
    const int g = chunk * nev + ev;
    const int stp = g / 6;
    const int s = g - stp * 6;
    const float sv = ((float)stp + kTsitC[s]) * 0.01f;
    const float bw = kTsitB[s];

    // stage h0 = relu(sv*u0 + gb0) as f16 into sact
    for (int i = tid; i < 4096; i += 256) {
      const int r = i >> 6, c4 = i & 63;
      const float4 uv = ((const float4*)u0)[(size_t)(row0 + r) * 64 + c4];
      const float4 bv = *(const float4*)(sb0 + c4 * 4);
      v4h h;
      h[0] = (_Float16)fmaxf(fmaf(sv, uv.x, bv.x), 0.0f);
      h[1] = (_Float16)fmaxf(fmaf(sv, uv.y, bv.y), 0.0f);
      h[2] = (_Float16)fmaxf(fmaf(sv, uv.z, bv.z), 0.0f);
      h[3] = (_Float16)fmaxf(fmaf(sv, uv.w, bv.w), 0.0f);
      *(v4h*)(sact + r * 264 + c4 * 4) = h;
    }
    __syncthreads();

    // GEMM1: d = h0 @ W1^T
    v8f d[2][4];
#pragma unroll
    for (int mt = 0; mt < 2; ++mt)
#pragma unroll
      for (int nt = 0; nt < 4; ++nt) d[mt][nt] = vzero8();
#pragma unroll
    for (int ks = 0; ks < 8; ++ks) {
      v16h af[2], bf[4];
#pragma unroll
      for (int mt = 0; mt < 2; ++mt)
        af[mt] = ld_fragA(sact + (wm * 32 + mt * 16 + lmod) * 264 + ks * 32 + lhalf * 8);
#pragma unroll
      for (int nt = 0; nt < 4; ++nt)
        bf[nt] = ld_fragB(sW1 + (wn * 64 + nt * 16 + lmod) * 264 + ks * 32 + lhalf * 8);
#pragma unroll
      for (int mt = 0; mt < 2; ++mt)
#pragma unroll
        for (int nt = 0; nt < 4; ++nt) d[mt][nt] = wmma16(af[mt], bf[nt], d[mt][nt]);
    }
    __syncthreads();  // everyone done reading h0 before overwrite

    // h1 = relu(d + gb1) back into sact (f16)
#pragma unroll
    for (int mt = 0; mt < 2; ++mt)
#pragma unroll
      for (int nt = 0; nt < 4; ++nt) {
        const int col = wn * 64 + nt * 16 + lmod;
        const float b = sb1[col];
#pragma unroll
        for (int v = 0; v < 8; ++v) {
          const int row = wm * 32 + mt * 16 + v + 8 * lhalf;
          sact[row * 264 + col] = (_Float16)fmaxf(d[mt][nt][v] + b, 0.0f);
        }
      }
    __syncthreads();

    // GEMM2: d = h1 @ W2^T ; accumulate b_s * relu(d + gb2)
#pragma unroll
    for (int mt = 0; mt < 2; ++mt)
#pragma unroll
      for (int nt = 0; nt < 4; ++nt) d[mt][nt] = vzero8();
#pragma unroll
    for (int ks = 0; ks < 8; ++ks) {
      v16h af[2], bf[4];
#pragma unroll
      for (int mt = 0; mt < 2; ++mt)
        af[mt] = ld_fragA(sact + (wm * 32 + mt * 16 + lmod) * 264 + ks * 32 + lhalf * 8);
#pragma unroll
      for (int nt = 0; nt < 4; ++nt)
        bf[nt] = ld_fragB(sW2 + (wn * 64 + nt * 16 + lmod) * 264 + ks * 32 + lhalf * 8);
#pragma unroll
      for (int mt = 0; mt < 2; ++mt)
#pragma unroll
        for (int nt = 0; nt < 4; ++nt) d[mt][nt] = wmma16(af[mt], bf[nt], d[mt][nt]);
    }
#pragma unroll
    for (int mt = 0; mt < 2; ++mt)
#pragma unroll
      for (int nt = 0; nt < 4; ++nt) {
        const int col = wn * 64 + nt * 16 + lmod;
        const float b = sb2[col];
#pragma unroll
        for (int v = 0; v < 8; ++v)
          aacc[mt][nt][v] += bw * fmaxf(d[mt][nt][v] + b, 0.0f);
      }
    __syncthreads();  // before next eval overwrites sact
  }

  float* ap = Apart + (size_t)chunk * (2048u * 256u);
#pragma unroll
  for (int mt = 0; mt < 2; ++mt)
#pragma unroll
    for (int nt = 0; nt < 4; ++nt) {
      const int col = wn * 64 + nt * 16 + lmod;
#pragma unroll
      for (int v = 0; v < 8; ++v) {
        const int row = row0 + wm * 32 + mt * 16 + v + 8 * lhalf;
        ap[(size_t)row * 256 + col] = aacc[mt][nt][v];
      }
    }
}

// ---------------------------------------------------------------------------
// Reduce eval-chunk partials -> A (f16 for the final GEMM)
__global__ void reduce_kernel(const float* __restrict__ Apart, int NC,
                              _Float16* __restrict__ Ah) {
  const size_t idx = (size_t)blockIdx.x * blockDim.x + threadIdx.x;
  if (idx >= (size_t)2048 * 256) return;
  float s = 0.0f;
  for (int c = 0; c < NC; ++c) s += Apart[(size_t)c * (2048u * 256u) + idx];
  Ah[idx] = (_Float16)s;
}

// ---------------------------------------------------------------------------
// Final: per workgroup (row-tile bm, output index i):
//   Q[64 rows][j=0..63] = A @ gW3[i*64+j,:]^T  (WMMA, K=256),
//   out[b,i] = f0[b,i] + sum_j x[b,j]*(DT*Q + gb3[i*64+j])
__global__ __launch_bounds__(128) void final_kernel(
    const _Float16* __restrict__ Ah, const _Float16* __restrict__ W3h,
    const float* __restrict__ x, const float* __restrict__ f0,
    const float* __restrict__ gb3, float* __restrict__ out) {
  __shared__ _Float16 sA[64 * 72];
  __shared__ _Float16 sW[64 * 72];
  __shared__ float sq[64 * 68];
  const int tid = threadIdx.x;
  const int bm = blockIdx.x * 64;
  const int ib = blockIdx.y;  // output feature i
  const int lane = tid & 31, wave = tid >> 5;
  const int wm = wave >> 1, wn = wave & 1;
  const int lmod = lane & 15, lhalf = lane >> 4;

  v8f d[2][2];
#pragma unroll
  for (int mt = 0; mt < 2; ++mt)
#pragma unroll
    for (int nt = 0; nt < 2; ++nt) d[mt][nt] = vzero8();

  for (int kc = 0; kc < 256; kc += 64) {
    for (int i = tid; i < 512; i += 128) {
      const int r = i >> 3, c = i & 7;
      *(uint4*)(sA + r * 72 + c * 8) =
          *(const uint4*)(Ah + (size_t)(bm + r) * 256 + kc + c * 8);
      *(uint4*)(sW + r * 72 + c * 8) =
          *(const uint4*)(W3h + (size_t)(ib * 64 + r) * 256 + kc + c * 8);
    }
    __syncthreads();
#pragma unroll
    for (int ks = 0; ks < 2; ++ks) {
      v16h af[2], bf[2];
#pragma unroll
      for (int mt = 0; mt < 2; ++mt)
        af[mt] = ld_fragA(sA + (wm * 32 + mt * 16 + lmod) * 72 + ks * 32 + lhalf * 8);
#pragma unroll
      for (int nt = 0; nt < 2; ++nt)
        bf[nt] = ld_fragB(sW + (wn * 32 + nt * 16 + lmod) * 72 + ks * 32 + lhalf * 8);
#pragma unroll
      for (int mt = 0; mt < 2; ++mt)
#pragma unroll
        for (int nt = 0; nt < 2; ++nt) d[mt][nt] = wmma16(af[mt], bf[nt], d[mt][nt]);
    }
    __syncthreads();
  }
  // stage Q tile to LDS
#pragma unroll
  for (int mt = 0; mt < 2; ++mt)
#pragma unroll
    for (int nt = 0; nt < 2; ++nt) {
      const int col = wn * 32 + nt * 16 + lmod;
#pragma unroll
      for (int v = 0; v < 8; ++v) {
        const int row = wm * 32 + mt * 16 + v + 8 * lhalf;
        sq[row * 68 + col] = d[mt][nt][v];
      }
    }
  __syncthreads();

  if (tid < 64) {
    const int r = tid;
    const float4* xr = (const float4*)(x + (size_t)(bm + r) * 64);
    const float4* gr = (const float4*)(gb3 + ib * 64);
    const float4* qr = (const float4*)(sq + r * 68);
    float acc = 0.0f;
#pragma unroll
    for (int j4 = 0; j4 < 16; ++j4) {
      const float4 xv = xr[j4];
      const float4 gv = gr[j4];
      const float4 qv = qr[j4];
      acc += xv.x * fmaf(0.01f, qv.x, gv.x);
      acc += xv.y * fmaf(0.01f, qv.y, gv.y);
      acc += xv.z * fmaf(0.01f, qv.z, gv.z);
      acc += xv.w * fmaf(0.01f, qv.w, gv.w);
    }
    out[(size_t)(bm + r) * 64 + ib] = f0[(size_t)(bm + r) * 64 + ib] + acc;
  }
}

// ---------------------------------------------------------------------------
extern "C" void kernel_launch(void* const* d_in, const int* in_sizes, int n_in,
                              void* d_out, int out_size, void* d_ws,
                              size_t ws_size, hipStream_t stream) {
  (void)in_sizes; (void)n_in; (void)out_size;
  const float* x   = (const float*)d_in[0];
  const float* iW0 = (const float*)d_in[1];
  const float* ib0 = (const float*)d_in[2];
  const float* iW1 = (const float*)d_in[3];
  const float* ib1 = (const float*)d_in[4];
  const float* iW2 = (const float*)d_in[5];
  const float* ib2 = (const float*)d_in[6];
  const float* iW3 = (const float*)d_in[7];
  const float* ib3 = (const float*)d_in[8];
  const float* gW0 = (const float*)d_in[9];
  const float* gb0 = (const float*)d_in[10];
  const float* gW1 = (const float*)d_in[11];
  const float* gb1 = (const float*)d_in[12];
  const float* gW2 = (const float*)d_in[13];
  const float* gb2 = (const float*)d_in[14];
  const float* gW3 = (const float*)d_in[15];
  const float* gb3 = (const float*)d_in[16];
  float* out = (float*)d_out;

  char* ws = (char*)d_ws;
  _Float16* xh    = (_Float16*)(ws + 0);        // 2048*64
  _Float16* iW0h  = (_Float16*)(ws + 262144);   // 256*64
  _Float16* iW1h  = (_Float16*)(ws + 294912);   // 256*256
  _Float16* iW2h  = (_Float16*)(ws + 425984);   // 256*256
  _Float16* iW3h  = (_Float16*)(ws + 557056);   // 64*256
  _Float16* gW0h  = (_Float16*)(ws + 589824);   // 256*64
  _Float16* gW1h  = (_Float16*)(ws + 622592);   // 256*256
  _Float16* gW2h  = (_Float16*)(ws + 753664);   // 256*256
  _Float16* gW3h  = (_Float16*)(ws + 884736);   // 4096*256
  _Float16* t0h   = (_Float16*)(ws + 2981888);  // 2048*256
  _Float16* t1h   = (_Float16*)(ws + 4030464);  // 2048*256
  float*    f0buf = (float*)   (ws + 5079040);  // 2048*64
  float*    u0buf = (float*)   (ws + 5603328);  // 2048*256
  _Float16* Ahbuf = (_Float16*)(ws + 7700480);  // 2048*256
  float*    Apart = (float*)   (ws + 8749056);  // NC * 2048*256 f32

  // weight / input conversions to f16
  auto cv = [&](const float* s, _Float16* d, int n) {
    cvt_kernel<<<(n + 255) / 256, 256, 0, stream>>>(s, d, n);
  };
  cv(x, xh, 131072);
  cv(iW0, iW0h, 16384);   cv(iW1, iW1h, 65536);
  cv(iW2, iW2h, 65536);   cv(iW3, iW3h, 16384);
  cv(gW0, gW0h, 16384);   cv(gW1, gW1h, 65536);
  cv(gW2, gW2h, 65536);   cv(gW3, gW3h, 1048576);

  const dim3 b128(128), b256(256);
  // init MLP: f0 = mlp(x)
  gemm_kernel<0><<<dim3(32, 4), b128, 0, stream>>>(xh,  iW0h, ib0, nullptr, t0h, 256, 64);
  gemm_kernel<0><<<dim3(32, 4), b128, 0, stream>>>(t0h, iW1h, ib1, nullptr, t1h, 256, 256);
  gemm_kernel<0><<<dim3(32, 4), b128, 0, stream>>>(t1h, iW2h, ib2, nullptr, t0h, 256, 256);
  gemm_kernel<1><<<dim3(32, 1), b128, 0, stream>>>(t0h, iW3h, ib3, f0buf, nullptr, 64, 256);
  // u0 = x @ gW0^T (no bias: bias enters per-stage as relu(sv*u0 + gb0))
  gemm_kernel<2><<<dim3(32, 4), b128, 0, stream>>>(xh, gW0h, nullptr, u0buf, nullptr, 256, 64);

  // pick eval-chunk count that fits the workspace (deterministic in ws_size)
  const size_t fixedB = 8749056;
  int NC = 1;
  const int opts[6] = {50, 25, 10, 5, 2, 1};
  for (int k = 0; k < 6; ++k) {
    if (fixedB + (size_t)opts[k] * 2097152u <= ws_size) { NC = opts[k]; break; }
  }
  const int CS = 100 / NC;  // RK steps per chunk

  // Phase B: 32 row-tiles x NC chunks; 307200 B dynamic LDS (< 320KB WGP LDS)
  phaseB_kernel<<<dim3(32, NC), b256, 307200, stream>>>(
      u0buf, gb0, gb1, gb2, gW1h, gW2h, Apart, CS);

  reduce_kernel<<<2048, 256, 0, stream>>>(Apart, NC, Ahbuf);

  final_kernel<<<dim3(32, 64), b128, 0, stream>>>(Ahbuf, gW3h, x, f0buf, gb3, out);
}
